// App_Enc_19937238188334
// MI455X (gfx1250) — matrix operand order repaired and checked by reading, not compiled, not run on hardware
//
#include <hip/hip_runtime.h>

typedef __attribute__((ext_vector_type(16))) _Float16 v16h;
typedef __attribute__((ext_vector_type(8)))  float    v8f;

#define B_   16
#define N_   32
#define K_   16
#define S_   256
#define SS_  (S_ * S_)
#define HPB_ 4              // h rows per block in kernel 1
#define G_   (S_ / HPB_)    // 64 h-groups per b

// ---------------------------------------------------------------------------
// Kernel 1: fused per-row softmax of raw_tps + partial app_vec GEMM via WMMA.
//   app_vec[b,n,k] = sum_{h,w} softmax_w(raw[b,n,h,:])[w] * x[b,k,h,w]
// One block per (b, group-of-4 h). Partial [32x16] written per group
// (no float atomics -> bitwise deterministic).
// ---------------------------------------------------------------------------
__global__ __launch_bounds__(256) void softmax_appvec_kernel(
    const float* __restrict__ x, const float* __restrict__ raw,
    float* __restrict__ partial)
{
    __shared__ _Float16 lds_p[N_][S_ + 16];   // softmax probs, f16, padded
    __shared__ _Float16 lds_x[K_][S_ + 16];   // x rows, f16, padded

    const int b    = blockIdx.x / G_;
    const int g    = blockIdx.x % G_;
    const int t    = threadIdx.x;
    const int wave = t >> 5;
    const int lane = t & 31;
    const int m    = lane & 15;
    const int half = lane >> 4;

    const float* rawb = raw + (size_t)b * N_ * SS_;
    const float* xb   = x   + (size_t)b * K_ * SS_;

    v8f acc = {};                 // per-wave (waves 0,1) 16x16 f32 accumulator

    for (int hh = 0; hh < HPB_; ++hh) {
        const int h = g * HPB_ + hh;

        // ---- stage 1: softmax rows (8 waves x 4 rows = 32 rows of 256) ----
        #pragma unroll
        for (int r = 0; r < 4; ++r) {
            const int n = wave * 4 + r;
            const float* row = rawb + (size_t)n * SS_ + (size_t)h * S_;
            float v[8];
            float mx = -__builtin_inff();
            #pragma unroll
            for (int e = 0; e < 8; ++e) {
                v[e] = row[e * 32 + lane];
                mx = fmaxf(mx, v[e]);
            }
            #pragma unroll
            for (int off = 16; off > 0; off >>= 1)
                mx = fmaxf(mx, __shfl_xor(mx, off));
            float s = 0.f;
            #pragma unroll
            for (int e = 0; e < 8; ++e) { v[e] = __expf(v[e] - mx); s += v[e]; }
            #pragma unroll
            for (int off = 16; off > 0; off >>= 1)
                s += __shfl_xor(s, off);
            const float inv = 1.0f / s;
            #pragma unroll
            for (int e = 0; e < 8; ++e)
                lds_p[n][e * 32 + lane] = (_Float16)(v[e] * inv);
        }
        // ---- stage x into LDS (8 waves x 2 rows = 16 rows) ----
        #pragma unroll
        for (int r = 0; r < 2; ++r) {
            const int k = wave * 2 + r;
            const float* row = xb + (size_t)k * SS_ + (size_t)h * S_;
            #pragma unroll
            for (int e = 0; e < 8; ++e)
                lds_x[k][e * 32 + lane] = (_Float16)row[e * 32 + lane];
        }
        __syncthreads();

        // ---- stage 2: D[n',k] += p[n',w] * x[k,w], wave0: n 0..15, wave1: 16..31
        if (wave < 2) {
            const int nrow = wave * 16 + m;   // this lane's A-matrix row
            #pragma unroll
            for (int wc = 0; wc < S_; wc += 32) {
                v16h a, bm;
                #pragma unroll
                for (int j = 0; j < 8; ++j) {
                    // A 16x32 f16 layout: lanes0-15 VGPR0-3:K0-7, VGPR4-7:K16-23
                    //                     lanes16-31 shift K by +8
                    const int ka = ((j < 4) ? 2 * j : 16 + 2 * (j - 4)) + half * 8;
                    a[2 * j]     = lds_p[nrow][wc + ka];
                    a[2 * j + 1] = lds_p[nrow][wc + ka + 1];
                    // B 32x16 f16 layout: same K order as A (K0-7,16-23; hi lanes +8)
                    const int kb = ((j < 4) ? 2 * j : 16 + 2 * (j - 4)) + half * 8;
                    bm[2 * j]     = lds_x[m][wc + kb];        // column k = m
                    bm[2 * j + 1] = lds_x[m][wc + kb + 1];
                }
                acc = __builtin_amdgcn_wmma_f32_16x16x32_f16(
                          false, a, false, bm, (short)0, acc, false, false);
            }
        }
        __syncthreads();   // protect LDS reuse next hh
    }

    // D layout: VGPR r -> lanes0-15: (M=r, N=lane), lanes16-31: (M=8+r, N=lane-16)
    if (wave < 2) {
        float* pb = partial + (((size_t)b * G_) + g) * (N_ * K_);
        #pragma unroll
        for (int r = 0; r < 8; ++r) {
            const int n = wave * 16 + r + half * 8;
            pb[n * K_ + m] = acc[r];
        }
    }
}

// ---------------------------------------------------------------------------
// Kernel 2: deterministic reduction of per-group partials -> app_vec[B,N,K]
// ---------------------------------------------------------------------------
__global__ __launch_bounds__(256) void reduce_appvec_kernel(
    const float* __restrict__ partial, float* __restrict__ app_vec)
{
    const int idx = blockIdx.x * 256 + threadIdx.x;   // [0, B*N*K)
    if (idx >= B_ * N_ * K_) return;
    const int b  = idx / (N_ * K_);
    const int nk = idx % (N_ * K_);
    const float* p = partial + (size_t)b * G_ * (N_ * K_) + nk;
    float s = 0.f;
    for (int g = 0; g < G_; ++g) s += p[(size_t)g * (N_ * K_)];
    app_vec[idx] = s;
}

// ---------------------------------------------------------------------------
// Kernel 3: out[b,k,h,w] = (sum_n fitted[b,n,h,w]*app_vec[b,n,k])
//                          / (1 + sum_n fitted[b,n,h,w])
// One v_wmma per 16-pixel x 16-K tile (K-dim = 32 = all n). Transpose through
// padded LDS so the [B,K,S,S] stores are coalesced.
// ---------------------------------------------------------------------------
__global__ __launch_bounds__(256) void combine_kernel(
    const float* __restrict__ fitted, const float* __restrict__ app_vec,
    float* __restrict__ out)
{
    __shared__ float lds_av[N_ * K_];          // app_vec[b] (512 f32)
    __shared__ float lds_dinv[128];            // 1/denom per local pixel
    __shared__ float lds_o[128][K_ + 1];       // stride-17 pad: conflict-free

    const int t    = threadIdx.x;
    const int wave = t >> 5;
    const int lane = t & 31;
    const int m    = lane & 15;
    const int half = lane >> 4;

    const int b    = blockIdx.x / (SS_ / 128);
    const int pix0 = (blockIdx.x % (SS_ / 128)) * 128;

    const float* avg = app_vec + (size_t)b * N_ * K_;
    for (int i = t; i < N_ * K_; i += 256) lds_av[i] = avg[i];
    __syncthreads();

    // ---- A fragment: fitted[b, n, pixel], 16 pixels per wave -------------
    const int pix = pix0 + wave * 16 + m;
    const float* fb = fitted + (size_t)b * N_ * SS_;
    v16h a;
    float dsum = 0.f;
    #pragma unroll
    for (int j = 0; j < 8; ++j) {
        const int n0 = ((j < 4) ? 2 * j : 16 + 2 * (j - 4)) + half * 8;
        const float f0 = fb[(size_t)n0 * SS_ + pix];
        const float f1 = fb[(size_t)(n0 + 1) * SS_ + pix];
        a[2 * j]     = (_Float16)f0;
        a[2 * j + 1] = (_Float16)f1;
        dsum += f0 + f1;                       // fp32 denominator path
    }
    dsum += __shfl_xor(dsum, 16);              // lanes L and L+16 cover all 32 n
    const float dinv = 1.0f / (1.0f + dsum);
    if (half == 0) lds_dinv[wave * 16 + m] = dinv;

    // ---- B fragment: app_vec[b, n, k], column k = m ----------------------
    v16h bm;
    #pragma unroll
    for (int j = 0; j < 8; ++j) {
        const int n0 = ((j < 4) ? 2 * j : 16 + 2 * (j - 4)) + half * 8;
        bm[2 * j]     = (_Float16)lds_av[n0 * K_ + m];
        bm[2 * j + 1] = (_Float16)lds_av[(n0 + 1) * K_ + m];
    }

    v8f acc = {};
    acc = __builtin_amdgcn_wmma_f32_16x16x32_f16(
              false, a, false, bm, (short)0, acc, false, false);

    __syncthreads();                           // lds_dinv visible to all
    #pragma unroll
    for (int r = 0; r < 8; ++r) {
        const int prow = wave * 16 + r + half * 8;   // local pixel in [0,128)
        lds_o[prow][m] = acc[r] * lds_dinv[prow];
    }
    __syncthreads();

    // ---- coalesced store: 2048 values, lanes sweep consecutive pixels ----
    float* ob = out + (size_t)b * K_ * SS_;
    #pragma unroll
    for (int i = 0; i < 8; ++i) {
        const int idx = i * 256 + t;
        const int k   = idx >> 7;              // idx / 128
        const int p   = idx & 127;
        ob[(size_t)k * SS_ + pix0 + p] = lds_o[p][k];
    }
}

// ---------------------------------------------------------------------------
extern "C" void kernel_launch(void* const* d_in, const int* in_sizes, int n_in,
                              void* d_out, int out_size, void* d_ws, size_t ws_size,
                              hipStream_t stream)
{
    const float* x      = (const float*)d_in[0];   // [B,K,S,S]
    const float* raw    = (const float*)d_in[1];   // [B,N,S,S]
    const float* fitted = (const float*)d_in[2];   // [B,N,S,S]
    float* out = (float*)d_out;                    // [B,K,S,S]

    // workspace: partial [B,G,N,K] (2 MB) then app_vec [B,N,K] (32 KB)
    float* partial = (float*)d_ws;
    float* app_vec = partial + (size_t)B_ * G_ * N_ * K_;

    softmax_appvec_kernel<<<B_ * G_, 256, 0, stream>>>(x, raw, partial);
    reduce_appvec_kernel<<<(B_ * N_ * K_ + 255) / 256, 256, 0, stream>>>(partial, app_vec);
    combine_kernel<<<B_ * (SS_ / 128), 256, 0, stream>>>(fitted, app_vec, out);
}
